// PhysGraphAttentionLayer_48086453846269
// MI455X (gfx1250) — matrix operand order repaired and checked by reading, not compiled, not run on hardware
//
#include <hip/hip_runtime.h>
#include <hip/hip_bf16.h>
#include <math.h>

// Graph attention layer for gfx1250 (MI455X).
// Dense GEMMs: v_wmma_f32_16x16x32_f16, LDS-free. Weights are pre-swizzled
// into a contiguous-k B-fragment order (this code's choice: lanes 0-15 hold K%32 in [0,16),
// lanes 16-31 hold K%32 in [16,32), slot = K&15), so each lane's B fragment
// is one contiguous 32B load. A fragments are two contiguous 16B chunks of a
// lane's own row in row-major f16 (OOB rows clamp to row 0; their outputs are
// masked at store, so no in-loop guard is needed). Workspace: ~210 MB.

typedef __attribute__((ext_vector_type(16))) _Float16 v16h;
typedef __attribute__((ext_vector_type(8)))  float    v8f;

#define DD 128

static __device__ __forceinline__ float warp_sum32(float x) {
#pragma unroll
  for (int off = 16; off > 0; off >>= 1) x += __shfl_xor(x, off, 32);
  return x;
}

static __device__ __forceinline__ void atomicMaxF(float* addr, float val) {
  // monotonic int-repr trick; buffer initialized to -inf
  if (val >= 0.0f) atomicMax((int*)addr, __float_as_int(val));
  else             atomicMin((unsigned int*)addr, (unsigned int)__float_as_int(val));
}

// B-fragment packed index for element (k, n) of a [K, Nc] matrix.
// Tile (k>>5, n>>4) is 512 halves; within it lane*16 + slot.
static __device__ __forceinline__ long bfrag_index(int k, int n, int nt4) {
  int kk   = k & 31;
  int lane = (n & 15) | (((kk >> 4) & 1) << 4);
  int slot = kk & 15;
  return ((long)(k >> 5) * nt4 + (n >> 4)) * 512 + lane * 16 + slot;
}

// ---------------------------------------------------------------- prep
__global__ void cvt_f16_kernel(const float* __restrict__ in, _Float16* __restrict__ out, long n) {
  long i = (long)blockIdx.x * blockDim.x + threadIdx.x;
  if (i < n) out[i] = (_Float16)in[i];
}

__global__ void pack_b_kernel(const float* __restrict__ W, _Float16* __restrict__ out,
                              int K, int Nc) {
  long idx = (long)blockIdx.x * blockDim.x + threadIdx.x;
  if (idx >= (long)K * Nc) return;
  int k = (int)(idx / Nc), n = (int)(idx % Nc);
  out[bfrag_index(k, n, Nc >> 4)] = (_Float16)W[idx];
}

__global__ void pack_qkvs_kernel(const float* __restrict__ Wq, const float* __restrict__ Wk,
                                 const float* __restrict__ Wv, const float* __restrict__ Ws,
                                 const float* __restrict__ bq, const float* __restrict__ bk,
                                 const float* __restrict__ bv, const float* __restrict__ bs,
                                 _Float16* __restrict__ W, float* __restrict__ bias) {
  int idx = blockIdx.x * blockDim.x + threadIdx.x;
  if (idx >= DD * 512) return;
  int k = idx >> 9, n = idx & 511;
  int sel = n >> 7, cc = n & 127;
  const float* w = sel == 0 ? Wq : sel == 1 ? Wk : sel == 2 ? Wv : Ws;
  W[bfrag_index(k, n, 32)] = (_Float16)w[k * DD + cc];
  if (idx < 512) {
    const float* b = sel == 0 ? bq : sel == 1 ? bk : sel == 2 ? bv : bs;
    bias[idx] = b[cc];
  }
}

__global__ void init_kernel(float* __restrict__ agg, float* __restrict__ maxs,
                            float* __restrict__ denom, long n128, int N) {
  long i = (long)blockIdx.x * blockDim.x + threadIdx.x;
  if (i < n128) agg[i] = 0.0f;
  if (i < N) { maxs[i] = -INFINITY; denom[i] = 0.0f; }
}

// ---------------------------------------------------------------- WMMA GEMM
// C[M,Nc] = A16[M,K] @ Bpacked[K,Nc] + bias ; act==1 -> exact GELU, f16 out.
// LDS-free: each wave owns a 16x64 strip. Grid (ceil(M/128), Nc/64), 8 waves.
__global__ __launch_bounds__(256)
void gemm_wmma_kernel(const _Float16* __restrict__ A, const _Float16* __restrict__ Bp,
                      const float* __restrict__ bias, float* __restrict__ Cf,
                      _Float16* __restrict__ Ch, int M, int K, int Nc, int act) {
  const int w    = threadIdx.x >> 5;
  const int lane = threadIdx.x & 31;
  const int hf   = lane >> 4;
  const int l16  = lane & 15;
  const long mrow = (long)blockIdx.x * 128 + w * 16 + l16;  // this lane's A row
  const int  nb0  = blockIdx.y * 4;                          // first 16-col subtile
  const int  nt4  = Nc >> 4;

  v8f acc[4];
#pragma unroll
  for (int n = 0; n < 4; ++n)
#pragma unroll
    for (int i = 0; i < 8; ++i) acc[n][i] = 0.0f;

  // OOB lanes clamp to row 0: their WMMA output rows are masked at store.
  const long mclamp = mrow < M ? mrow : 0;
  const _Float16* arow = A + mclamp * (long)K;

  for (int k0 = 0; k0 < K; k0 += 32) {
    // A fragment (ISA 7.12.2, 16-bit A 16x32): two contiguous 16B chunks.
    union { uint4 u[2]; v16h v; } au;
    au.u[0] = *(const uint4*)(arow + k0 + hf * 16);
    au.u[1] = *(const uint4*)(arow + k0 + hf * 16 + 8);
    v16h afrag = au.v;

    const _Float16* btile = Bp + ((long)(k0 >> 5) * nt4 + nb0) * 512 + lane * 16;
#pragma unroll
    for (int n = 0; n < 4; ++n) {
      v16h bfrag = *(const v16h*)(btile + n * 512);  // contiguous 32B per lane
      acc[n] = __builtin_amdgcn_wmma_f32_16x16x32_f16(
          false, afrag, false, bfrag, (short)0, acc[n], false, false);
    }
  }

  // store: acc[n][r] -> (row = w*16 + hf*8 + r, col = n*16 + l16)
#pragma unroll
  for (int n = 0; n < 4; ++n) {
    int col = blockIdx.y * 64 + n * 16 + l16;
    float bv = bias ? bias[col] : 0.0f;
#pragma unroll
    for (int r = 0; r < 8; ++r) {
      long row = (long)blockIdx.x * 128 + w * 16 + hf * 8 + r;
      if (row < M) {
        float x = acc[n][r] + bv;
        if (act == 1) {
          float g = 0.5f * x * (1.0f + erff(x * 0.70710678118654752f));
          Ch[row * Nc + col] = (_Float16)g;
        } else {
          Cf[row * Nc + col] = x;
        }
      }
    }
  }
}

// ---------------------------------------------------------------- edge phase
// qkvs row layout: [q(0:128) | k(128:256) | v(256:384) | selfpre(384:512)]
__global__ __launch_bounds__(256)
void edge_score_max_kernel(const int* __restrict__ src, const int* __restrict__ dst,
                           const float* __restrict__ qkvs, float* __restrict__ score,
                           float* __restrict__ maxs, int E) {
  int e = blockIdx.x * 8 + (threadIdx.x >> 5);
  int lane = threadIdx.x & 31;
  if (e >= E) return;
  int s = src[e], d = dst[e];
  float4 qv = *(const float4*)(qkvs + (long)d * 512 + lane * 4);
  float4 kv = *(const float4*)(qkvs + (long)s * 512 + 128 + lane * 4);
  float p = qv.x * kv.x + qv.y * kv.y + qv.z * kv.z + qv.w * kv.w;
  p = warp_sum32(p);
  if (lane == 0) {
    float sc = p * 0.08838834764831845f;  // 128^-0.5
    score[e] = sc;
    atomicMaxF(&maxs[d], sc);
  }
}

__global__ void edge_exp_kernel(const int* __restrict__ dst, float* __restrict__ score,
                                const float* __restrict__ maxs, float* __restrict__ denom, int E) {
  int e = blockIdx.x * blockDim.x + threadIdx.x;
  if (e >= E) return;
  int d = dst[e];
  float ex = __expf(score[e] - maxs[d]);
  score[e] = ex;  // score buffer now holds exp_s
  atomicAdd(&denom[d], ex);
}

__global__ __launch_bounds__(256)
void edge_agg_kernel(const int* __restrict__ src, const int* __restrict__ dst,
                     const float* __restrict__ qkvs, const float* __restrict__ score,
                     const float* __restrict__ denom, float* __restrict__ agg, int E) {
  int e = blockIdx.x * 8 + (threadIdx.x >> 5);
  int lane = threadIdx.x & 31;
  if (e >= E) return;
  int s = src[e], d = dst[e];
  float alpha = score[e] / fmaxf(denom[d], 1e-12f);
  float4 vv = *(const float4*)(qkvs + (long)s * 512 + 256 + lane * 4);
  float* ap = agg + (long)d * DD + lane * 4;
  atomicAdd(ap + 0, vv.x * alpha);
  atomicAdd(ap + 1, vv.y * alpha);
  atomicAdd(ap + 2, vv.z * alpha);
  atomicAdd(ap + 3, vv.w * alpha);
}

// ---------------------------------------------------------------- layernorms
__global__ __launch_bounds__(256)
void h1_ln_kernel(const float* __restrict__ h, const float* __restrict__ qkvs,
                  const float* __restrict__ aggout, const float* __restrict__ g1,
                  const float* __restrict__ be1, float* __restrict__ h1,
                  _Float16* __restrict__ h1h, int N) {
  int row = blockIdx.x * 8 + (threadIdx.x >> 5);
  int lane = threadIdx.x & 31;
  if (row >= N) return;
  float x[4], sum = 0.f, sq = 0.f;
#pragma unroll
  for (int t = 0; t < 4; ++t) {
    int j = lane * 4 + t;
    float pre = qkvs[(long)row * 512 + 384 + j] + aggout[(long)row * DD + j];
    float xi = h[(long)row * DD + j] + fmaxf(pre, 0.0f);
    x[t] = xi; sum += xi; sq += xi * xi;
  }
  sum = warp_sum32(sum); sq = warp_sum32(sq);
  float mean = sum * (1.0f / 128.0f);
  float var  = sq * (1.0f / 128.0f) - mean * mean;
  float inv  = rsqrtf(var + 1e-5f);
#pragma unroll
  for (int t = 0; t < 4; ++t) {
    int j = lane * 4 + t;
    float y = (x[t] - mean) * inv * g1[j] + be1[j];
    h1[(long)row * DD + j]  = y;
    h1h[(long)row * DD + j] = (_Float16)y;
  }
}

__global__ __launch_bounds__(256)
void ln2_kernel(const float* __restrict__ h1, const float* __restrict__ ffn2,
                const float* __restrict__ g2, const float* __restrict__ be2,
                float* __restrict__ out, int N) {
  int row = blockIdx.x * 8 + (threadIdx.x >> 5);
  int lane = threadIdx.x & 31;
  if (row >= N) return;
  float x[4], sum = 0.f, sq = 0.f;
#pragma unroll
  for (int t = 0; t < 4; ++t) {
    int j = lane * 4 + t;
    float xi = h1[(long)row * DD + j] + ffn2[(long)row * DD + j];
    x[t] = xi; sum += xi; sq += xi * xi;
  }
  sum = warp_sum32(sum); sq = warp_sum32(sq);
  float mean = sum * (1.0f / 128.0f);
  float var  = sq * (1.0f / 128.0f) - mean * mean;
  float inv  = rsqrtf(var + 1e-5f);
#pragma unroll
  for (int t = 0; t < 4; ++t) {
    int j = lane * 4 + t;
    out[(long)row * DD + j] = (x[t] - mean) * inv * g2[j] + be2[j];
  }
}

// ---------------------------------------------------------------- launch
extern "C" void kernel_launch(void* const* d_in, const int* in_sizes, int n_in,
                              void* d_out, int out_size, void* d_ws, size_t ws_size,
                              hipStream_t stream) {
  const float* h     = (const float*)d_in[0];
  const int*   eidx  = (const int*)d_in[1];
  const float* Wself = (const float*)d_in[2];
  const float* bself = (const float*)d_in[3];
  const float* Wq    = (const float*)d_in[4];
  const float* bq    = (const float*)d_in[5];
  const float* Wk    = (const float*)d_in[6];
  const float* bk    = (const float*)d_in[7];
  const float* Wv    = (const float*)d_in[8];
  const float* bv    = (const float*)d_in[9];
  const float* Wout  = (const float*)d_in[10];
  const float* bout  = (const float*)d_in[11];
  const float* g1    = (const float*)d_in[12];
  const float* be1   = (const float*)d_in[13];
  const float* W1    = (const float*)d_in[14];
  const float* b1    = (const float*)d_in[15];
  const float* W2    = (const float*)d_in[16];
  const float* b2    = (const float*)d_in[17];
  const float* g2    = (const float*)d_in[18];
  const float* be2   = (const float*)d_in[19];

  const long N = in_sizes[0] / DD;
  const long E = in_sizes[1] / 2;
  const int* esrc = eidx;
  const int* edst = eidx + E;

  // workspace layout (256B aligned slices)
  char* ws = (char*)d_ws;
  size_t off = 0;
  auto take = [&](size_t bytes) -> char* {
    char* p = ws + off;
    off = (off + bytes + 255) & ~(size_t)255;
    return p;
  };
  _Float16* h16    = (_Float16*)take(N * DD * 2);        // later reused as agg16
  _Float16* wqkvs  = (_Float16*)take(DD * 512 * 2);      // packed B-frag layout
  float*    bqkvs  = (float*)take(512 * 4);
  _Float16* wout16 = (_Float16*)take(DD * DD * 2);       // packed
  _Float16* w1_16  = (_Float16*)take(DD * 512 * 2);      // packed
  _Float16* w2_16  = (_Float16*)take(512 * DD * 2);      // packed
  float*    qkvs   = (float*)take(N * 512 * 4);          // later reused as ffn16 (f16)
  float*    agg    = (float*)take(N * DD * 4);           // later reused as ffn2 out
  float*    aggout = (float*)take(N * DD * 4);
  float*    h1     = (float*)take(N * DD * 4);
  _Float16* h1_16  = (_Float16*)take(N * DD * 2);
  float*    score  = (float*)take(E * 4);
  float*    maxs   = (float*)take(N * 4);
  float*    denom  = (float*)take(N * 4);
  (void)ws_size;
  _Float16* agg16   = h16;                 // alias: h16 dead after QKVS GEMM
  _Float16* ffn16   = (_Float16*)qkvs;     // alias: qkvs dead after h1
  float*    ffn2out = agg;                 // alias: agg dead after aggout GEMM

  const long n128 = N * DD;
  const int mb = (int)((N + 127) / 128);

  // 1) precision prep + weight packing into WMMA fragment order
  cvt_f16_kernel<<<(int)((n128 + 255) / 256), 256, 0, stream>>>(h, h16, n128);
  pack_qkvs_kernel<<<(DD * 512 + 255) / 256, 256, 0, stream>>>(
      Wq, Wk, Wv, Wself, bq, bk, bv, bself, wqkvs, bqkvs);
  pack_b_kernel<<<(DD * DD + 255) / 256, 256, 0, stream>>>(Wout, wout16, DD, DD);
  pack_b_kernel<<<(DD * 512 + 255) / 256, 256, 0, stream>>>(W1, w1_16, DD, 512);
  pack_b_kernel<<<(512 * DD + 255) / 256, 256, 0, stream>>>(W2, w2_16, 512, DD);
  init_kernel<<<(int)((n128 + 255) / 256), 256, 0, stream>>>(agg, maxs, denom, n128, (int)N);

  // 2) fused QKV+self GEMM: [N,128] x [128,512]
  gemm_wmma_kernel<<<dim3(mb, 8), 256, 0, stream>>>(
      h16, wqkvs, bqkvs, qkvs, nullptr, (int)N, DD, 512, 0);

  // 3) edge attention (scatter softmax + aggregate)
  edge_score_max_kernel<<<(int)((E + 7) / 8), 256, 0, stream>>>(esrc, edst, qkvs, score, maxs, (int)E);
  edge_exp_kernel<<<(int)((E + 255) / 256), 256, 0, stream>>>(edst, score, maxs, denom, (int)E);
  edge_agg_kernel<<<(int)((E + 7) / 8), 256, 0, stream>>>(esrc, edst, qkvs, score, denom, agg, (int)E);

  // 4) agg @ W_out + b_out
  cvt_f16_kernel<<<(int)((n128 + 255) / 256), 256, 0, stream>>>(agg, agg16, n128);
  gemm_wmma_kernel<<<dim3(mb, 2), 256, 0, stream>>>(
      agg16, wout16, bout, aggout, nullptr, (int)N, DD, DD, 0);

  // 5) h1 = LN(h + relu(selfpre + aggout))
  h1_ln_kernel<<<(int)((N + 7) / 8), 256, 0, stream>>>(h, qkvs, aggout, g1, be1, h1, h1_16, (int)N);

  // 6) FFN: gelu(h1@W1+b1) (f16) then @W2+b2 (f32)
  gemm_wmma_kernel<<<dim3(mb, 8), 256, 0, stream>>>(
      h1_16, w1_16, b1, nullptr, ffn16, (int)N, DD, 512, 1);
  gemm_wmma_kernel<<<dim3(mb, 2), 256, 0, stream>>>(
      ffn16, w2_16, b2, ffn2out, nullptr, (int)N, 512, DD, 0);

  // 7) out = LN(h1 + ffn)
  ln2_kernel<<<(int)((N + 7) / 8), 256, 0, stream>>>(h1, ffn2out, g2, be2, (float*)d_out, (int)N);
}
